// Qwen2MoeSparseMLP_50835232916119
// MI455X (gfx1250) — matrix operand order repaired and checked by reading, not compiled, not run on hardware
//
#include <hip/hip_runtime.h>

// Problem constants (match reference)
#define T_TOK 2048
#define H_DIM 2048
#define E_NUM 8
#define I_DIM 1408
#define TILE_M 32                 // tokens per expert tile (2 x 16-row A frags)

#define KC1 (H_DIM / 32)          // 64  k-chunks in GEMM1
#define KC2 (I_DIM / 32)          // 44  k-chunks in GEMM2
#define NT1 (I_DIM / 16)          // 88  n-tiles in GEMM1
#define NT2 (H_DIM / 16)          // 128 n-tiles in GEMM2

typedef __attribute__((ext_vector_type(16))) __bf16 v16bf;
typedef __attribute__((ext_vector_type(8)))  float  v8f;

union Frag { v16bf v; uint4 q[2]; };

__device__ __forceinline__ unsigned f2bf(float f) {
    union { float f; unsigned u; } c{f};
    unsigned u = c.u;
    unsigned r = u + 0x7FFFu + ((u >> 16) & 1u);   // round-to-nearest-even
    return r >> 16;
}

// ---------------------------------------------------------------- cnt zero
__global__ void moe_zero_cnt(int* __restrict__ cnt) {
    if (threadIdx.x < E_NUM) cnt[threadIdx.x] = 0;
}

// ---------------------------------------------------------------- x -> bf16
__global__ void moe_pack_x(const float* __restrict__ x, unsigned* __restrict__ xb) {
    int i = blockIdx.x * blockDim.x + threadIdx.x;        // over T*H/2 dwords
    if (i < T_TOK * H_DIM / 2) {
        float a = x[2 * i], b = x[2 * i + 1];
        xb[i] = f2bf(a) | (f2bf(b) << 16);
    }
}

// ------------------------------------------------- weights -> frag-tiled bf16
// src: E x [K,N] fp32 row-major. dst dword layout: [e][nt][kc][lane][j=0..7]
// lane 0-15  -> col = nt*16 + lane,    K = kc*32 + {2j,2j+1}
// lane 16-31 -> col = nt*16 + lane-16, K = kc*32 + 16 + {2j,2j+1}
__global__ void moe_pack_w(const float* __restrict__ src, unsigned* __restrict__ dst,
                           int K, int N) {
    int gid = blockIdx.x * blockDim.x + threadIdx.x;      // E*(N/16)*(K/32)*32 threads
    int kcN = K >> 5, ntN = N >> 4;
    int total = E_NUM * ntN * kcN * 32;
    if (gid >= total) return;
    int lane = gid & 31;
    int rest = gid >> 5;
    int kc = rest % kcN;  rest /= kcN;
    int nt = rest % ntN;  int e = rest / ntN;

    int col   = nt * 16 + (lane & 15);
    int kbase = kc * 32 + ((lane >> 4) << 4);
    const float* s = src + (size_t)e * K * N + (size_t)kbase * N + col;

    unsigned o[8];
#pragma unroll
    for (int j = 0; j < 8; ++j) {
        unsigned lo = f2bf(s[(size_t)(2 * j)     * N]);
        unsigned hi = f2bf(s[(size_t)(2 * j + 1) * N]);
        o[j] = lo | (hi << 16);
    }
    size_t frag = (((size_t)e * ntN + nt) * kcN + kc) * 32 + lane;
    uint4* d = (uint4*)(dst + frag * 8);
    d[0] = make_uint4(o[0], o[1], o[2], o[3]);
    d[1] = make_uint4(o[4], o[5], o[6], o[7]);
}

// ---------------------------------------------------------------- router
__global__ __launch_bounds__(256) void moe_router_kernel(
        const float* __restrict__ x, const float* __restrict__ gw,
        int* __restrict__ cnt, int* __restrict__ tokslot, float* __restrict__ tokw) {
    const int wave = threadIdx.x >> 5;
    const int lane = threadIdx.x & 31;
    const int t = blockIdx.x * 8 + wave;
    if (t >= T_TOK) return;

    float acc[E_NUM];
#pragma unroll
    for (int e = 0; e < E_NUM; ++e) acc[e] = 0.f;
    const float* xrow = x + (size_t)t * H_DIM;
    for (int h = lane; h < H_DIM; h += 32) {
        float xv = xrow[h];
        const float* g = gw + (size_t)h * E_NUM;
#pragma unroll
        for (int e = 0; e < E_NUM; ++e) acc[e] += xv * g[e];
    }
#pragma unroll
    for (int e = 0; e < E_NUM; ++e)
#pragma unroll
        for (int off = 16; off > 0; off >>= 1)
            acc[e] += __shfl_xor(acc[e], off, 32);

    if (lane == 0) {
        float m = acc[0];
#pragma unroll
        for (int e = 1; e < E_NUM; ++e) m = fmaxf(m, acc[e]);
        float p[E_NUM]; float s = 0.f;
#pragma unroll
        for (int e = 0; e < E_NUM; ++e) { p[e] = __expf(acc[e] - m); s += p[e]; }
        float inv = __builtin_amdgcn_rcpf(s);
#pragma unroll
        for (int e = 0; e < E_NUM; ++e) p[e] *= inv;

        int i0 = 0;
#pragma unroll
        for (int e = 1; e < E_NUM; ++e) if (p[e] > p[i0]) i0 = e;
        int i1 = (i0 == 0) ? 1 : 0;
#pragma unroll
        for (int e = 0; e < E_NUM; ++e) if (e != i0 && p[e] > p[i1]) i1 = e;

        float w0 = p[i0], w1v = p[i1];
        float dinv = __builtin_amdgcn_rcpf(w0 + w1v);
        int s0 = atomicAdd(&cnt[i0], 1);
        tokslot[i0 * T_TOK + s0] = t * 2;      tokw[i0 * T_TOK + s0] = w0 * dinv;
        int s1 = atomicAdd(&cnt[i1], 1);
        tokslot[i1 * T_TOK + s1] = t * 2 + 1;  tokw[i1 * T_TOK + s1] = w1v * dinv;
    }
}

// ------------------------------------------------------- pipelined K panel
// 2-deep software pipeline: wmma for chunk kc consumes registers loaded at
// kc-2, so the compiler can use non-zero wait counts instead of draining.
__device__ __forceinline__ void gemm_panel(
        const unsigned short* abase, int astride,
        const unsigned* bbase, int kcN,
        v8f& acc0, v8f& acc1) {
    Frag b[2], a0[2], a1[2];

#define LOADK(slot, kc)                                                        \
    do {                                                                       \
        const uint4* bp_ = (const uint4*)(bbase + (size_t)(kc) * 256);         \
        b[slot].q[0] = bp_[0];  b[slot].q[1] = bp_[1];                         \
        const unsigned short* ap_ = abase + (kc) * 32;                         \
        a0[slot].q[0] = *(const uint4*)(ap_);                                  \
        a0[slot].q[1] = *(const uint4*)(ap_ + 8);                              \
        a1[slot].q[0] = *(const uint4*)(ap_ + 16 * astride);                   \
        a1[slot].q[1] = *(const uint4*)(ap_ + 16 * astride + 8);               \
    } while (0)

    LOADK(0, 0);
    LOADK(1, 1);
#pragma unroll 2
    for (int kc = 0; kc < kcN - 2; ++kc) {          // kcN even -> clean unroll
        int cur = kc & 1;
        acc0 = __builtin_amdgcn_wmma_f32_16x16x32_bf16(
            false, a0[cur].v, false, b[cur].v, (short)0, acc0, false, false);
        acc1 = __builtin_amdgcn_wmma_f32_16x16x32_bf16(
            false, a1[cur].v, false, b[cur].v, (short)0, acc1, false, false);
        LOADK(cur, kc + 2);                          // prefetch 2 chunks ahead
    }
    // epilogue: chunks kcN-2 (slot 0) and kcN-1 (slot 1)
    acc0 = __builtin_amdgcn_wmma_f32_16x16x32_bf16(
        false, a0[0].v, false, b[0].v, (short)0, acc0, false, false);
    acc1 = __builtin_amdgcn_wmma_f32_16x16x32_bf16(
        false, a1[0].v, false, b[0].v, (short)0, acc1, false, false);
    acc0 = __builtin_amdgcn_wmma_f32_16x16x32_bf16(
        false, a0[1].v, false, b[1].v, (short)0, acc0, false, false);
    acc1 = __builtin_amdgcn_wmma_f32_16x16x32_bf16(
        false, a1[1].v, false, b[1].v, (short)0, acc1, false, false);
#undef LOADK
}

// ---------------------------------------------------------------- expert MLP
// Block = (expert, 32-token tile). 8 waves, 221 KB LDS (CDNA5-only footprint).
__global__ __launch_bounds__(256) void moe_expert_kernel(
        const unsigned* __restrict__ xb,
        const unsigned* __restrict__ w1b, const unsigned* __restrict__ w2b,
        const int* __restrict__ cnt, const int* __restrict__ tokslot,
        const float* __restrict__ tokw, float* __restrict__ ybuf) {
    extern __shared__ char smem[];
    int*            toks = (int*)smem;                       // 128 B
    float*          wts  = (float*)(smem + 128);             // 128 B
    unsigned short* Xs   = (unsigned short*)(smem + 256);                       // 128 KB
    unsigned short* Mid  = (unsigned short*)(smem + 256 + TILE_M * H_DIM * 2);  // 88 KB

    const int e    = blockIdx.x >> 6;          // T_TOK/TILE_M == 64 tiles
    const int tile = blockIdx.x & 63;
    const int base = tile * TILE_M;
    const int n_e  = cnt[e];
    if (base >= n_e) return;                   // uniform per block

    const int tid = threadIdx.x;
    if (tid < TILE_M) {
        int m = base + tid;
        toks[tid] = (m < n_e) ? tokslot[e * T_TOK + m] : -1;
        wts[tid]  = (m < n_e) ? tokw[e * T_TOK + m]   : 0.f;
    }
    __syncthreads();

    // Gather 32 pre-converted bf16 token rows -> LDS (b128 copies)
    for (int idx = tid; idx < TILE_M * (H_DIM / 8); idx += 256) {
        int m = idx >> 8;                       // H_DIM/8 == 256 uint4 per row
        int c = idx & 255;
        int ts = toks[m];
        uint4 v = make_uint4(0u, 0u, 0u, 0u);
        if (ts >= 0)
            v = ((const uint4*)xb)[(size_t)(ts >> 1) * (H_DIM / 8) + c];
        ((uint4*)Xs)[(size_t)m * (H_DIM / 8) + c] = v;
    }
    __syncthreads();

    const int wave = tid >> 5;
    const int lane = tid & 31;
    const int lhal = lane >> 4;
    const int l15  = lane & 15;
    const unsigned* w1e = w1b + (size_t)e * NT1 * KC1 * 256;   // 256 dwords per frag
    const unsigned* w2e = w2b + (size_t)e * NT2 * KC2 * 256;

    // ---- Phase 1: Mid = silu(Xs @ w1[e])   (32 x 1408, K = 2048)
    for (int nt = wave; nt < NT1; nt += 8) {
        const int col = nt * 16 + l15;
        v8f acc0 = {}, acc1 = {};
        gemm_panel(Xs + l15 * H_DIM + lhal * 16, H_DIM,
                   w1e + ((size_t)nt * KC1 * 32 + lane) * 8, KC1, acc0, acc1);
#pragma unroll
        for (int r = 0; r < 8; ++r) {
            int row = r + lhal * 8;
            float v0 = acc0[r];
            Mid[row * I_DIM + col] =
                (unsigned short)f2bf(v0 * __builtin_amdgcn_rcpf(1.f + __expf(-v0)));
            float v1 = acc1[r];
            Mid[(row + 16) * I_DIM + col] =
                (unsigned short)f2bf(v1 * __builtin_amdgcn_rcpf(1.f + __expf(-v1)));
        }
    }
    __syncthreads();

    // ---- Phase 2: ybuf[slot] = rowweight * (Mid @ w2[e])   (32 x 2048, K = 1408)
    for (int nt = wave; nt < NT2; nt += 8) {
        const int col = nt * 16 + l15;
        v8f acc0 = {}, acc1 = {};
        gemm_panel(Mid + l15 * I_DIM + lhal * 16, I_DIM,
                   w2e + ((size_t)nt * KC2 * 32 + lane) * 8, KC2, acc0, acc1);
#pragma unroll
        for (int r = 0; r < 8; ++r) {
            int row = r + lhal * 8;
            int ts0 = toks[row];
            if (ts0 >= 0)
                ybuf[(size_t)ts0 * H_DIM + col] = acc0[r] * wts[row];
            int ts1 = toks[row + 16];
            if (ts1 >= 0)
                ybuf[(size_t)ts1 * H_DIM + col] = acc1[r] * wts[row + 16];
        }
    }
}

// ---------------------------------------------------------------- combine
__global__ void moe_combine_kernel(const float* __restrict__ ybuf,
                                   float* __restrict__ out) {
    int i = blockIdx.x * blockDim.x + threadIdx.x;        // over T*H/4 float4
    if (i >= T_TOK * H_DIM / 4) return;
    int t  = i >> 9;                                       // H/4 == 512
    int h4 = i & 511;
    const float4* y = (const float4*)ybuf;
    float4 a = y[(size_t)(t * 2)     * 512 + h4];
    float4 b = y[(size_t)(t * 2 + 1) * 512 + h4];
    float4 r; r.x = a.x + b.x; r.y = a.y + b.y; r.z = a.z + b.z; r.w = a.w + b.w;
    ((float4*)out)[i] = r;
}

// ---------------------------------------------------------------- launch
extern "C" void kernel_launch(void* const* d_in, const int* in_sizes, int n_in,
                              void* d_out, int out_size, void* d_ws, size_t ws_size,
                              hipStream_t stream) {
    const float* x   = (const float*)d_in[0];
    const float* gw  = (const float*)d_in[1];
    const float* w1  = (const float*)d_in[2];
    const float* w2  = (const float*)d_in[3];
    float* out = (float*)d_out;

    // workspace layout (all 16B-aligned)
    char* ws = (char*)d_ws;
    int*      cnt  = (int*)ws;                                         // 64 B
    int*      tsl  = (int*)(ws + 64);                                  // 64 KB
    float*    tkw  = (float*)(ws + 64 + 65536);                        // 64 KB
    unsigned* xb   = (unsigned*)(ws + 131200);                         // 8 MB  bf16 x
    unsigned* w1b  = (unsigned*)(ws + 131200 + 8388608);               // 46 MB bf16 w1
    unsigned* w2b  = (unsigned*)(ws + 131200 + 8388608 + 46137344);    // 46 MB bf16 w2
    float*    ybuf = (float*)(ws + 131200 + 8388608 + 2 * 46137344);   // 32 MB contributions

    moe_zero_cnt<<<1, 64, 0, stream>>>(cnt);
    moe_pack_x<<<(T_TOK * H_DIM / 2 + 255) / 256, 256, 0, stream>>>(x, xb);
    {   // w1: K=H, N=I ; w2: K=I, N=H
        int thr1 = E_NUM * NT1 * KC1 * 32;
        moe_pack_w<<<(thr1 + 255) / 256, 256, 0, stream>>>(w1, w1b, H_DIM, I_DIM);
        int thr2 = E_NUM * NT2 * KC2 * 32;
        moe_pack_w<<<(thr2 + 255) / 256, 256, 0, stream>>>(w2, w2b, I_DIM, H_DIM);
    }
    moe_router_kernel<<<T_TOK / 8, 256, 0, stream>>>(x, gw, cnt, tsl, tkw);

    const size_t lds_bytes = 256 + (size_t)TILE_M * H_DIM * 2 + (size_t)TILE_M * I_DIM * 2;
    moe_expert_kernel<<<E_NUM * (T_TOK / TILE_M), 256, lds_bytes, stream>>>(
        xb, w1b, w2b, cnt, tsl, tkw, ybuf);

    moe_combine_kernel<<<(T_TOK * H_DIM / 4 + 255) / 256, 256, 0, stream>>>(ybuf, out);
}
